// PointNetPTMSG_91156385890399
// MI455X (gfx1250) — matrix operand order repaired and checked by reading, not compiled, not run on hardware
//
#include <hip/hip_runtime.h>
#include <cstdint>
#include <cstddef>

#define BIGF 1e10f

static inline int cp32(int x){ return (x+31)&~31; }

typedef __attribute__((ext_vector_type(16))) _Float16 v16h;
typedef __attribute__((ext_vector_type(8)))  _Float16 v8h;
typedef __attribute__((ext_vector_type(8)))  float    v8f;

// ---------------------------------------------------------------------------
// xyz extraction + piece ids (4 equal pieces)
// ---------------------------------------------------------------------------
__global__ void extract_xyz_piece(const float* __restrict__ x, float* __restrict__ xyz,
                                  int* __restrict__ piece, int N, int inCh, int perPiece) {
    int i = blockIdx.x * blockDim.x + threadIdx.x;
    if (i >= N) return;
    xyz[i*3+0] = x[(size_t)i*inCh+0];
    xyz[i*3+1] = x[(size_t)i*inCh+1];
    xyz[i*3+2] = x[(size_t)i*inCh+2];
    piece[i] = i / perPiece;
}

// ---------------------------------------------------------------------------
// Farthest point sampling: single block, sequential over S steps, parallel argmax
// ---------------------------------------------------------------------------
#define FPS_BS 256
__global__ void fps_kernel(const float* __restrict__ xyz, int N, int S,
                           int* __restrict__ fidx, float* __restrict__ dist) {
    __shared__ float sval[FPS_BS];
    __shared__ int   sidx[FPS_BS];
    const int tid = threadIdx.x;
    for (int i = tid; i < N; i += FPS_BS) dist[i] = BIGF;
    __syncthreads();
    int far = 0;
    for (int s = 0; s < S; ++s) {
        if (tid == 0) fidx[s] = far;
        float cx = xyz[far*3+0], cy = xyz[far*3+1], cz = xyz[far*3+2];
        float bv = -1.0f; int bi = 0x7fffffff;
        for (int i = tid; i < N; i += FPS_BS) {
            float dx = xyz[i*3+0]-cx, dy = xyz[i*3+1]-cy, dz = xyz[i*3+2]-cz;
            float d = dx*dx + dy*dy + dz*dz;
            float nd = fminf(dist[i], d);
            dist[i] = nd;
            if (nd > bv) { bv = nd; bi = i; }   // strided i increasing: first-max kept
        }
        sval[tid] = bv; sidx[tid] = bi;
        __syncthreads();
        for (int off = FPS_BS/2; off > 0; off >>= 1) {
            if (tid < off) {
                float ov = sval[tid+off]; int oi = sidx[tid+off];
                if (ov > sval[tid] || (ov == sval[tid] && oi < sidx[tid])) {
                    sval[tid] = ov; sidx[tid] = oi;
                }
            }
            __syncthreads();
        }
        far = sidx[0];
        __syncthreads();
    }
}

__global__ void gather_centers(const float* __restrict__ xyz, const int* __restrict__ piece,
                               const int* __restrict__ fidx, int S,
                               float* __restrict__ nxyz, int* __restrict__ npiece) {
    int s = blockIdx.x * blockDim.x + threadIdx.x;
    if (s >= S) return;
    int j = fidx[s];
    nxyz[s*3+0] = xyz[j*3+0];
    nxyz[s*3+1] = xyz[j*3+1];
    nxyz[s*3+2] = xyz[j*3+2];
    npiece[s] = piece[j];
}

// ---------------------------------------------------------------------------
// Ball query: first K same-piece neighbors (index order) within radius; pad with first
// ---------------------------------------------------------------------------
__global__ void ball_query(const float* __restrict__ nxyz, const int* __restrict__ npiece,
                           const float* __restrict__ xyz, const int* __restrict__ piece,
                           int S, int N, float r2, int K, int* __restrict__ idx) {
    int s = blockIdx.x * blockDim.x + threadIdx.x;
    if (s >= S) return;
    float cx = nxyz[s*3+0], cy = nxyz[s*3+1], cz = nxyz[s*3+2];
    int p = npiece[s];
    int cnt = 0;
    for (int j = 0; j < N && cnt < K; ++j) {
        if (piece[j] != p) continue;
        float dx = xyz[j*3+0]-cx, dy = xyz[j*3+1]-cy, dz = xyz[j*3+2]-cz;
        if (dx*dx + dy*dy + dz*dz <= r2) idx[(size_t)s*K + cnt++] = j;
    }
    int f = (cnt > 0) ? idx[(size_t)s*K] : 0;
    for (; cnt < K; ++cnt) idx[(size_t)s*K + cnt] = f;
}

// ---------------------------------------------------------------------------
// Group gather: Hin[row] = concat(points[idx], xyz[idx]-center), f16, zero-padded K
// ---------------------------------------------------------------------------
template <typename T>
__global__ void sa_gather(const T* __restrict__ pts, int srcStride, int Cpts,
                          const float* __restrict__ xyz, const float* __restrict__ nxyz,
                          const int* __restrict__ idx, int S, int K, int CinPad,
                          _Float16* __restrict__ Hin) {
    int row = blockIdx.x * blockDim.x + threadIdx.x;
    if (row >= S * K) return;
    int s = row / K;
    int j = idx[row];
    _Float16* o = Hin + (size_t)row * CinPad;
    for (int c = 0; c < Cpts; ++c) o[c] = (_Float16)(float)pts[(size_t)j*srcStride + c];
    o[Cpts+0] = (_Float16)(xyz[j*3+0] - nxyz[s*3+0]);
    o[Cpts+1] = (_Float16)(xyz[j*3+1] - nxyz[s*3+1]);
    o[Cpts+2] = (_Float16)(xyz[j*3+2] - nxyz[s*3+2]);
    for (int c = Cpts+3; c < CinPad; ++c) o[c] = (_Float16)0.f;
}

// ---------------------------------------------------------------------------
// Pack f32 weights [Cout,Cin] -> f16 [NPad,KPad] zero-padded; pad bias too
// ---------------------------------------------------------------------------
__global__ void pack_w(const float* __restrict__ W, const float* __restrict__ b,
                       int Cout, int Cin, int NPad, int KPad,
                       _Float16* __restrict__ Wp, float* __restrict__ bp) {
    int t = blockIdx.x * blockDim.x + threadIdx.x;
    int tot = NPad * KPad;
    if (t < tot) {
        int n = t / KPad, k = t % KPad;
        Wp[t] = (n < Cout && k < Cin) ? (_Float16)W[(size_t)n*Cin + k] : (_Float16)0.f;
    }
    if (t < NPad) bp[t] = (t < Cout) ? b[t] : 0.f;
}

// ---------------------------------------------------------------------------
// WMMA GEMM: C = relu(A @ W^T + b). f16 in, f32 acc, f16 out.
// Block = 8 waves (256 thr). Block computes a 256-row x 32-col patch:
//   - 32-col weight panel async-staged into LDS (GLOBAL_LOAD_ASYNC_TO_LDS_B128,
//     s_wait_asynccnt, then ds_load_b128 fragment reads)
//   - each wave owns 32 rows x 32 cols = 2x2 WMMA tiles (4 accumulators),
//     A streamed from global (b128) — 4 WMMA per 4 A-loads + 4 LDS B-loads.
// M padded to 32; out-of-range waves clamp A rows (EXEC stays all-1) and skip stores.
// ---------------------------------------------------------------------------
__global__ __launch_bounds__(256) void wmma_gemm_relu(
    const _Float16* __restrict__ A, const _Float16* __restrict__ W,
    const float* __restrict__ bias, _Float16* __restrict__ C,
    int KPad, int NPad, int M32) {
    extern __shared__ _Float16 smem[];   // [32 x KPad] weight panel
    const int lane = threadIdx.x & 31;
    const int wid  = threadIdx.x >> 5;
    const int l15  = lane & 15;
    const int hs   = lane >> 4;

    // ---- async stage of 32-row weight panel (rows blockIdx.y*32 ..) into LDS ----
    {
        const char* wsrc = (const char*)(W + (size_t)blockIdx.y * 32 * KPad);
        unsigned lbase = (unsigned)(uintptr_t)(void*)smem;
        int chunks = (32 * KPad * 2) >> 4;               // 16B chunks
        for (int c = threadIdx.x; c < chunks; c += 256) {
            unsigned laddr = lbase + ((unsigned)c << 4);
            unsigned long long gaddr = (unsigned long long)(uintptr_t)(wsrc + ((size_t)c << 4));
            asm volatile("global_load_async_to_lds_b128 %0, %1, off"
                         :: "v"(laddr), "v"(gaddr) : "memory");
        }
        asm volatile("s_wait_asynccnt 0x0" ::: "memory");
        __syncthreads();
    }

    const int rowBase = blockIdx.x * 256 + wid * 32;
    const bool active = rowBase < M32;
    const int rb = active ? rowBase : 0;                 // clamp: keep loads in-bounds
    const _Float16* a0p = A + ((size_t)rb + l15)      * KPad;
    const _Float16* a1p = A + ((size_t)rb + 16 + l15) * KPad;
    const _Float16* b0p = smem + (size_t)l15        * KPad;
    const _Float16* b1p = smem + (size_t)(16 + l15) * KPad;

    v8f acc00 = {}, acc01 = {}, acc10 = {}, acc11 = {};
    for (int k = 0; k < KPad; k += 32) {
        __builtin_prefetch(a0p + k + 256, 0, 0);
        __builtin_prefetch(a1p + k + 256, 0, 0);
        // A fragments (16x32 f16): halves 0-7 -> K=k+hs*8.., halves 8-15 -> K=k+16+hs*8..
        v8h a0lo = *(const v8h*)(a0p + k + hs*8);
        v8h a0hi = *(const v8h*)(a0p + k + 16 + hs*8);
        v8h a1lo = *(const v8h*)(a1p + k + hs*8);
        v8h a1hi = *(const v8h*)(a1p + k + 16 + hs*8);
        // B fragments (32x16 f16) from LDS: halves 0-7 -> K=k+hs*8.., halves 8-15 -> K=k+16+hs*8.. for col l15
        v8h b0lo = *(const v8h*)(b0p + k + hs*8);
        v8h b0hi = *(const v8h*)(b0p + k + 16 + hs*8);
        v8h b1lo = *(const v8h*)(b1p + k + hs*8);
        v8h b1hi = *(const v8h*)(b1p + k + 16 + hs*8);
        v16h a0 = __builtin_shufflevector(a0lo, a0hi, 0,1,2,3,4,5,6,7,8,9,10,11,12,13,14,15);
        v16h a1 = __builtin_shufflevector(a1lo, a1hi, 0,1,2,3,4,5,6,7,8,9,10,11,12,13,14,15);
        v16h b0 = __builtin_shufflevector(b0lo, b0hi, 0,1,2,3,4,5,6,7,8,9,10,11,12,13,14,15);
        v16h b1 = __builtin_shufflevector(b1lo, b1hi, 0,1,2,3,4,5,6,7,8,9,10,11,12,13,14,15);
        acc00 = __builtin_amdgcn_wmma_f32_16x16x32_f16(false, a0, false, b0, (short)0, acc00, false, false);
        acc01 = __builtin_amdgcn_wmma_f32_16x16x32_f16(false, a0, false, b1, (short)0, acc01, false, false);
        acc10 = __builtin_amdgcn_wmma_f32_16x16x32_f16(false, a1, false, b0, (short)0, acc10, false, false);
        acc11 = __builtin_amdgcn_wmma_f32_16x16x32_f16(false, a1, false, b1, (short)0, acc11, false, false);
    }

    if (active) {            // wave-uniform branch: EXEC untouched within wave
        const int cb = blockIdx.y * 32;
        const float bb0 = bias[cb + l15];
        const float bb1 = bias[cb + 16 + l15];
#pragma unroll
        for (int r = 0; r < 8; ++r) {
            size_t row0 = (size_t)rowBase + r + 8*hs;
            size_t row1 = (size_t)rowBase + 16 + r + 8*hs;
            float v;
            v = acc00[r] + bb0; v = v > 0.f ? v : 0.f;
            C[row0 * NPad + cb + l15]      = (_Float16)v;
            v = acc01[r] + bb1; v = v > 0.f ? v : 0.f;
            C[row0 * NPad + cb + 16 + l15] = (_Float16)v;
            v = acc10[r] + bb0; v = v > 0.f ? v : 0.f;
            C[row1 * NPad + cb + l15]      = (_Float16)v;
            v = acc11[r] + bb1; v = v > 0.f ? v : 0.f;
            C[row1 * NPad + cb + 16 + l15] = (_Float16)v;
        }
    }
}

// Final layer variant: single wave per 16x16 tile, f32 out, no relu, narrow N (13 cols)
__global__ __launch_bounds__(32) void wmma_gemm_f32(
    const _Float16* __restrict__ A, const _Float16* __restrict__ W,
    const float* __restrict__ bias, float* __restrict__ C,
    int KPad, int Ncols, int ldc, int M) {
    const int lane = threadIdx.x;
    const int l15  = lane & 15;
    const int hs   = lane >> 4;
    const _Float16* a = A + ((size_t)blockIdx.x * 16 + l15) * KPad;
    const _Float16* w = W + ((size_t)blockIdx.y * 16 + l15) * KPad;
    v8f acc = {};
    for (int k = 0; k < KPad; k += 32) {
        v8h alo = *(const v8h*)(a + k + hs*8);
        v8h ahi = *(const v8h*)(a + k + 16 + hs*8);
        v8h blo = *(const v8h*)(w + k + hs*8);
        v8h bhi = *(const v8h*)(w + k + 16 + hs*8);
        v16h av = __builtin_shufflevector(alo, ahi, 0,1,2,3,4,5,6,7,8,9,10,11,12,13,14,15);
        v16h bv = __builtin_shufflevector(blo, bhi, 0,1,2,3,4,5,6,7,8,9,10,11,12,13,14,15);
        acc = __builtin_amdgcn_wmma_f32_16x16x32_f16(false, av, false, bv,
                                                     (short)0, acc, false, false);
    }
    const float bb = bias[blockIdx.y*16 + l15];
    const int col = blockIdx.y*16 + l15;
#pragma unroll
    for (int r = 0; r < 8; ++r) {
        int row = blockIdx.x*16 + r + 8*hs;
        if (row < M && col < Ncols) C[(size_t)row*ldc + col] = acc[r] + bb;
    }
}

// ---------------------------------------------------------------------------
// Max pool over group dim K; write into concat slice of level features
// ---------------------------------------------------------------------------
__global__ void maxpool_kernel(const _Float16* __restrict__ act, int S, int K,
                               int C, int Cpad, _Float16* __restrict__ out,
                               int outStride, int outOff) {
    int t = blockIdx.x * blockDim.x + threadIdx.x;
    if (t >= S * C) return;
    int s = t / C, c = t % C;
    float m = -3.4e38f;
    for (int k = 0; k < K; ++k)
        m = fmaxf(m, (float)act[((size_t)s*K + k) * Cpad + c]);
    out[(size_t)s*outStride + outOff + c] = (_Float16)m;
}

// ---------------------------------------------------------------------------
// Feature propagation: 3-NN (same piece, else BIG) inverse-distance interp, concat skip
// ---------------------------------------------------------------------------
__global__ void fp_interp_kernel(const float* __restrict__ xyz1, const int* __restrict__ p1,
                                 const float* __restrict__ xyz2, const int* __restrict__ p2,
                                 const _Float16* __restrict__ f2, int C2, int C2s,
                                 const _Float16* __restrict__ f1, int C1, int C1s,
                                 int N1, int N2, int CinPad, _Float16* __restrict__ Hin) {
    int i = blockIdx.x * blockDim.x + threadIdx.x;
    if (i >= N1) return;
    float cx = xyz1[i*3+0], cy = xyz1[i*3+1], cz = xyz1[i*3+2];
    int pp = p1[i];
    float d0 = 3.4e38f, d1 = 3.4e38f, d2v = 3.4e38f;
    int   i0 = 0, i1 = 0, i2 = 0;
    for (int j = 0; j < N2; ++j) {
        float d;
        if (p2[j] != pp) d = BIGF;
        else {
            float dx = xyz2[j*3+0]-cx, dy = xyz2[j*3+1]-cy, dz = xyz2[j*3+2]-cz;
            d = dx*dx + dy*dy + dz*dz;
        }
        if (d < d2v) {
            if (d < d1) {
                if (d < d0) { d2v=d1; i2=i1; d1=d0; i1=i0; d0=d; i0=j; }
                else        { d2v=d1; i2=i1; d1=d;  i1=j; }
            } else          { d2v=d;  i2=j; }
        }
    }
    float w0 = 1.f/(d0+1e-8f), w1 = 1.f/(d1+1e-8f), w2 = 1.f/(d2v+1e-8f);
    float s = w0+w1+w2; w0/=s; w1/=s; w2/=s;
    _Float16* o = Hin + (size_t)i * CinPad;
    int off = 0;
    if (f1) { for (int c = 0; c < C1; ++c) o[c] = f1[(size_t)i*C1s + c]; off = C1; }
    const _Float16* a0 = f2 + (size_t)i0*C2s;
    const _Float16* a1 = f2 + (size_t)i1*C2s;
    const _Float16* a2 = f2 + (size_t)i2*C2s;
    for (int c = 0; c < C2; ++c)
        o[off+c] = (_Float16)(w0*(float)a0[c] + w1*(float)a1[c] + w2*(float)a2[c]);
    for (int c = off + C2; c < CinPad; ++c) o[c] = (_Float16)0.f;
}

// ---------------------------------------------------------------------------
// Host-side orchestration helpers
// ---------------------------------------------------------------------------
struct Bufs {
    _Float16 *Hin, *actA, *actB, *Wp;
    float* bp;
    int* nidx;
    float* dist;
    int* fidx;
};

static void run_mlp(hipStream_t st, void* const* d_in, int& pi,
                    const _Float16* A0, int rows, const int* dims, int nl,
                    Bufs& B, _Float16* dest) {
    const int M32 = cp32(rows);
    const _Float16* cur = A0;
    for (int l = 0; l < nl; ++l) {
        const float* W = (const float*)d_in[pi++];
        const float* b = (const float*)d_in[pi++];
        int Cin = dims[l], Cout = dims[l+1];
        int KPad = cp32(Cin), NPad = cp32(Cout);
        int tot = NPad * KPad;
        pack_w<<<(tot+255)/256, 256, 0, st>>>(W, b, Cout, Cin, NPad, KPad, B.Wp, B.bp);
        _Float16* out = (l == nl-1) ? dest : ((l & 1) ? B.actB : B.actA);
        dim3 g((M32 + 255)/256, NPad/32);
        size_t sh = (size_t)32 * KPad * sizeof(_Float16);
        wmma_gemm_relu<<<g, 256, sh, st>>>(cur, B.Wp, B.bp, out, KPad, NPad, M32);
        cur = out;
    }
}

static void run_sa(hipStream_t st, void* const* d_in, int& pi,
                   const float* xyzIn, const int* pieceIn, int Nin,
                   const void* ptsIn, int ptsIsF32, int ptsStride, int Cpts,
                   int S, const float* radii, const int* Ks,
                   const int dims0[4], const int dims1[4],
                   float* nxyz, int* npiece, _Float16* outF, int Ctot, Bufs& B) {
    fps_kernel<<<1, FPS_BS, 0, st>>>(xyzIn, Nin, S, B.fidx, B.dist);
    gather_centers<<<(S+255)/256, 256, 0, st>>>(xyzIn, pieceIn, B.fidx, S, nxyz, npiece);
    const int* dimsArr[2] = { dims0, dims1 };
    int off = 0;
    for (int br = 0; br < 2; ++br) {
        int K = Ks[br]; float r = radii[br];
        ball_query<<<(S+255)/256, 256, 0, st>>>(nxyz, npiece, xyzIn, pieceIn, S, Nin, r*r, K, B.nidx);
        int rows = S * K;
        int CinPad = cp32(Cpts + 3);
        if (ptsIsF32)
            sa_gather<float><<<(rows+255)/256, 256, 0, st>>>(
                (const float*)ptsIn, ptsStride, Cpts, xyzIn, nxyz, B.nidx, S, K, CinPad, B.Hin);
        else
            sa_gather<_Float16><<<(rows+255)/256, 256, 0, st>>>(
                (const _Float16*)ptsIn, ptsStride, Cpts, xyzIn, nxyz, B.nidx, S, K, CinPad, B.Hin);
        const int* dm = dimsArr[br];
        run_mlp(st, d_in, pi, B.Hin, rows, dm, 3, B, B.actA);
        int Cout = dm[3];
        maxpool_kernel<<<((S*Cout)+255)/256, 256, 0, st>>>(B.actA, S, K, Cout, cp32(Cout),
                                                          outF, Ctot, off);
        off += Cout;
    }
}

static void run_fp(hipStream_t st, void* const* d_in, int& pi,
                   const float* xyz1, const int* p1, int N1,
                   const float* xyz2, const int* p2, int N2,
                   const _Float16* f1, int C1, int C1s,
                   const _Float16* f2, int C2, int C2s,
                   const int* dims, int nl, Bufs& B, _Float16* dest) {
    int CinPad = cp32(C1 + C2);
    fp_interp_kernel<<<(N1+255)/256, 256, 0, st>>>(xyz1, p1, xyz2, p2, f2, C2, C2s,
                                                   f1, C1, C1s, N1, N2, CinPad, B.Hin);
    run_mlp(st, d_in, pi, B.Hin, N1, dims, nl, B, dest);
}

// ---------------------------------------------------------------------------
extern "C" void kernel_launch(void* const* d_in, const int* in_sizes, int n_in,
                              void* d_out, int out_size, void* d_ws, size_t ws_size,
                              hipStream_t stream) {
    (void)n_in; (void)out_size; (void)ws_size;
    const float* x = (const float*)d_in[0];
    const int N0 = in_sizes[0] / 9;
    const int S1 = (int)(0.15 * N0);   // 2457
    const int S2 = (int)(0.25 * S1);   // 614
    const int S3 = (int)(0.25 * S2);   // 153
    const int S4 = (int)(0.25 * S3);   // 38

    // ---- workspace layout ----
    char* wp = (char*)d_ws;
    auto alloc = [&](size_t bytes) -> char* {
        char* p = wp; wp += (bytes + 255) & ~(size_t)255; return p;
    };
    float* xyz0  = (float*)alloc((size_t)N0*3*4);
    int*   piece0= (int*)  alloc((size_t)N0*4);
    float* dist  = (float*)alloc((size_t)N0*4);
    int*   fidx  = (int*)  alloc((size_t)S1*4);
    float* xyz1  = (float*)alloc((size_t)S1*3*4);  int* piece1 = (int*)alloc((size_t)S1*4);
    float* xyz2  = (float*)alloc((size_t)S2*3*4);  int* piece2 = (int*)alloc((size_t)S2*4);
    float* xyz3  = (float*)alloc((size_t)S3*3*4);  int* piece3 = (int*)alloc((size_t)S3*4);
    float* xyz4  = (float*)alloc((size_t)S4*3*4);  int* piece4 = (int*)alloc((size_t)S4*4);
    int*   nidx  = (int*)  alloc((size_t)S1*32*4);
    _Float16* l1f  = (_Float16*)alloc((size_t)S1*96*2);
    _Float16* l2f  = (_Float16*)alloc((size_t)S2*256*2);
    _Float16* l3f  = (_Float16*)alloc((size_t)S3*512*2);
    _Float16* l4f  = (_Float16*)alloc((size_t)S4*1024*2);
    _Float16* l3f2 = (_Float16*)alloc((size_t)cp32(S3)*256*2);
    _Float16* l2f2 = (_Float16*)alloc((size_t)cp32(S2)*256*2);
    _Float16* l1f2 = (_Float16*)alloc((size_t)cp32(S1)*128*2);
    _Float16* f0   = (_Float16*)alloc((size_t)N0*128*2);
    _Float16* Hin  = (_Float16*)alloc((size_t)3000000*2);
    _Float16* actA = (_Float16*)alloc((size_t)5300000*2);
    _Float16* actB = (_Float16*)alloc((size_t)5300000*2);
    _Float16* Wp   = (_Float16*)alloc((size_t)420000*2);
    float*    bp   = (float*)   alloc((size_t)512*4);

    Bufs B{ Hin, actA, actB, Wp, bp, nidx, dist, fidx };
    int pi = 1;  // param cursor: x is d_in[0]; 68 (W,b) tensors follow, then batch_length

    extract_xyz_piece<<<(N0+255)/256, 256, 0, stream>>>(x, xyz0, piece0, N0, 9, N0/4);

    static const int k2[2] = { 16, 32 };
    static const int sa1d0[4] = { 12, 16, 16, 32 },   sa1d1[4] = { 12, 32, 32, 64 };
    static const int sa2d0[4] = { 99, 64, 64, 128 },  sa2d1[4] = { 99, 64, 96, 128 };
    static const int sa3d [4] = { 259, 128, 196, 256 };
    static const int sa4d0[4] = { 515, 256, 256, 512 }, sa4d1[4] = { 515, 256, 384, 512 };
    const float r1[2] = { 0.05f, 0.1f };
    const float r2[2] = { 0.1f,  0.2f };
    const float r3[2] = { 0.2f,  0.4f };
    const float r4[2] = { 0.4f,  0.8f };

    run_sa(stream, d_in, pi, xyz0, piece0, N0, x,   1, 9,   9,   S1, r1, k2, sa1d0, sa1d1, xyz1, piece1, l1f, 96,  B);
    run_sa(stream, d_in, pi, xyz1, piece1, S1, l1f, 0, 96,  96,  S2, r2, k2, sa2d0, sa2d1, xyz2, piece2, l2f, 256, B);
    run_sa(stream, d_in, pi, xyz2, piece2, S2, l2f, 0, 256, 256, S3, r3, k2, sa3d,  sa3d,  xyz3, piece3, l3f, 512, B);
    run_sa(stream, d_in, pi, xyz3, piece3, S3, l3f, 0, 512, 512, S4, r4, k2, sa4d0, sa4d1, xyz4, piece4, l4f, 1024, B);

    static const int fp4d[3] = { 1536, 256, 256 };
    static const int fp3d[3] = { 512,  256, 256 };
    static const int fp2d[3] = { 352,  256, 128 };
    static const int fp1d[4] = { 128, 128, 128, 128 };

    run_fp(stream, d_in, pi, xyz3, piece3, S3, xyz4, piece4, S4, l3f, 512, 512, l4f,  1024, 1024, fp4d, 2, B, l3f2);
    run_fp(stream, d_in, pi, xyz2, piece2, S2, xyz3, piece3, S3, l2f, 256, 256, l3f2, 256,  256,  fp3d, 2, B, l2f2);
    run_fp(stream, d_in, pi, xyz1, piece1, S1, xyz2, piece2, S2, l1f, 96,  96,  l2f2, 256,  256,  fp2d, 2, B, l1f2);
    run_fp(stream, d_in, pi, xyz0, piece0, N0, xyz1, piece1, S1, nullptr, 0, 0, l1f2, 128,  128,  fp1d, 3, B, f0);

    // conv1: [128 -> 13], no relu, f32 out to d_out
    const float* Wc = (const float*)d_in[pi++];
    const float* bc = (const float*)d_in[pi++];
    pack_w<<<(16*128 + 255)/256, 256, 0, stream>>>(Wc, bc, 13, 128, 16, 128, B.Wp, B.bp);
    wmma_gemm_f32<<<dim3(N0/16, 1), 32, 0, stream>>>(f0, B.Wp, B.bp, (float*)d_out,
                                                     128, 13, 13, N0);
}
